// TRANSFORMER_41635412967956
// MI455X (gfx1250) — matrix operand order repaired and checked by reading, not compiled, not run on hardware
//
#include <hip/hip_runtime.h>

typedef __attribute__((ext_vector_type(16))) __bf16 v16bf;
typedef __attribute__((ext_vector_type(8)))  float  v8f;

namespace {
constexpr int   kN     = 50000;
constexpr int   kE     = 400000;
constexpr int   kH     = 4;
constexpr int   kHD    = 64;
constexpr int   kED    = 16;
constexpr int   kG     = 128;
constexpr int   kDN    = 256;
constexpr float kScale = 0.25f;              // 1/sqrt(D), D=16
constexpr unsigned kEncNegInf = 0x007FFFFFu; // monotone encoding of -inf
}

static __device__ __forceinline__ unsigned fenc(float f) {
  unsigned u = __float_as_uint(f);
  return u ^ (unsigned)(((int)u >> 31) | 0x80000000);
}
static __device__ __forceinline__ float fdec(unsigned u) {
  unsigned b = (u & 0x80000000u) ? (u ^ 0x80000000u) : ~u;
  return __uint_as_float(b);
}
static __device__ __forceinline__ __bf16 f2bf(float f) {
  unsigned u = __float_as_uint(f);
  unsigned short s = (unsigned short)((u + 0x7FFFu + ((u >> 16) & 1u)) >> 16);
  return __builtin_bit_cast(__bf16, s);
}

// ---------------------------------------------------------------------------
// Per-layer init: agg = 0, m = enc(-inf), s = 0
// ---------------------------------------------------------------------------
__global__ __launch_bounds__(256) void tc_init_layer(float* __restrict__ agg,
                                                     unsigned* __restrict__ m_u,
                                                     float* __restrict__ ssum) {
  int t = blockIdx.x * 256 + threadIdx.x;
  if (t < kN * kHD) agg[t] = 0.0f;
  if (t < kN * kH) { m_u[t] = kEncNegInf; ssum[t] = 0.0f; }
}

// ---------------------------------------------------------------------------
// q/k/v/skip = X @ W + b   (bf16 WMMA, f32 accumulate)
// 8 waves/block, each wave owns one 16-row tile; 4 weights x 4 col-tiles x 2 K-frags
// ---------------------------------------------------------------------------
__global__ __launch_bounds__(256) void tc_gemm_qkvs(
    const float* __restrict__ X,
    const float* __restrict__ Wq, const float* __restrict__ bq,
    const float* __restrict__ Wk, const float* __restrict__ bk,
    const float* __restrict__ Wv, const float* __restrict__ bv,
    const float* __restrict__ Ws, const float* __restrict__ bs,
    float* __restrict__ Q, float* __restrict__ K,
    float* __restrict__ V, float* __restrict__ S) {
  __shared__ __bf16 sW[4][kHD * kHD];   // 32 KB: four 64x64 weights, bf16
  __shared__ __bf16 sA[8][16 * kHD];    // 16 KB: one 16x64 tile per wave

  const int tid  = threadIdx.x;
  const int lane = tid & 31;
  const int wave = tid >> 5;

  const float* Wsrc[4] = {Wq, Wk, Wv, Ws};
  #pragma unroll
  for (int w = 0; w < 4; ++w)
    for (int i = tid; i < kHD * kHD; i += 256) sW[w][i] = f2bf(Wsrc[w][i]);

  const int tile   = blockIdx.x * 8 + wave;
  const bool active = (tile * 16) < kN;
  if (active) {
    const float* xrow = X + (size_t)tile * 16 * kHD;
    for (int i = lane; i < 16 * kHD; i += 32) sA[wave][i] = f2bf(xrow[i]);
  }
  __syncthreads();
  if (!active) return;

  // A fragments: lane<16 -> M=lane, K halves {0..7,16..23}; lanes 16..31 shifted by 8
  const int mr = lane & 15;
  const int kb = (lane < 16) ? 0 : 8;
  v16bf a0, a1;
  #pragma unroll
  for (int i = 0; i < 16; ++i) {
    int kk = kb + (i < 8 ? i : i + 8);
    a0[i] = sA[wave][mr * kHD + kk];
    a1[i] = sA[wave][mr * kHD + 32 + kk];
  }

  const float* Bias[4] = {bq, bk, bv, bs};
  float* Out[4] = {Q, K, V, S};
  const int col16  = lane & 15;
  const int rowoff = (lane < 16) ? 0 : 8;
  const int krow   = (lane < 16) ? 0 : 8;

  #pragma unroll
  for (int w = 0; w < 4; ++w) {
    #pragma unroll
    for (int nt = 0; nt < 4; ++nt) {
      const int n0 = nt * 16;
      v16bf b0, b1;
      #pragma unroll
      for (int i = 0; i < 16; ++i) {
        b0[i] = sW[w][(krow + (i < 8 ? i : i + 8)) * kHD + n0 + col16];
        b1[i] = sW[w][(32 + krow + (i < 8 ? i : i + 8)) * kHD + n0 + col16];
      }
      const float bias = Bias[w][n0 + col16];
      v8f acc;
      #pragma unroll
      for (int r = 0; r < 8; ++r) acc[r] = bias;
      acc = __builtin_amdgcn_wmma_f32_16x16x32_bf16(false, a0, false, b0,
                                                    (short)0, acc, false, false);
      acc = __builtin_amdgcn_wmma_f32_16x16x32_bf16(false, a1, false, b1,
                                                    (short)0, acc, false, false);
      float* out = Out[w];
      #pragma unroll
      for (int r = 0; r < 8; ++r)
        out[(size_t)(tile * 16 + rowoff + r) * kHD + n0 + col16] = acc[r];
    }
  }
}

// ---------------------------------------------------------------------------
// Pass A: per (edge, head): logit = scale * q[dst] . (k[src] + edge_attr@We)
//         atomic segment-max into m (monotone uint encoding)
// ---------------------------------------------------------------------------
__global__ __launch_bounds__(256) void tc_edge_logits(
    const float* __restrict__ EA, const int* __restrict__ src,
    const int* __restrict__ dst, const float* __restrict__ We,
    const float* __restrict__ Q, const float* __restrict__ K,
    float* __restrict__ logits, unsigned* __restrict__ m_u) {
  int t = blockIdx.x * 256 + threadIdx.x;
  if (t >= kE * kH) return;
  const int e  = t >> 2;
  const int hh = t & 3;
  const int sv = src[e];
  const int dv = dst[e];

  float ef[16];
  #pragma unroll
  for (int d = 0; d < 16; ++d) ef[d] = 0.0f;
  const float* ea = EA + (size_t)e * kED;
  #pragma unroll
  for (int j = 0; j < kED; ++j) {
    const float aj = ea[j];
    const float* wr = We + j * kHD + hh * 16;
    #pragma unroll
    for (int d = 0; d < 16; ++d) ef[d] += aj * wr[d];
  }

  const float* qr = Q + (size_t)dv * kHD + hh * 16;
  const float* kr = K + (size_t)sv * kHD + hh * 16;
  float lg = 0.0f;
  #pragma unroll
  for (int d = 0; d < 16; ++d) lg += qr[d] * (kr[d] + ef[d]);
  lg *= kScale;

  logits[t] = lg;
  atomicMax(&m_u[dv * kH + hh], fenc(lg));
}

// ---------------------------------------------------------------------------
// Pass B: p = exp(logit - m[dst]); segment-sum into s
// ---------------------------------------------------------------------------
__global__ __launch_bounds__(256) void tc_edge_softmax(
    const int* __restrict__ dst, const unsigned* __restrict__ m_u,
    float* __restrict__ pbuf, float* __restrict__ ssum) {
  int t = blockIdx.x * 256 + threadIdx.x;
  if (t >= kE * kH) return;
  const int e  = t >> 2;
  const int hh = t & 3;
  const int dv = dst[e];
  const float p = expf(pbuf[t] - fdec(m_u[dv * kH + hh]));
  pbuf[t] = p;
  atomicAdd(&ssum[dv * kH + hh], p);
}

// ---------------------------------------------------------------------------
// Pass C: a = p / (s + eps);  agg[dst] += a * (v[src] + edge_attr@We)
// ---------------------------------------------------------------------------
__global__ __launch_bounds__(256) void tc_edge_agg(
    const float* __restrict__ EA, const int* __restrict__ src,
    const int* __restrict__ dst, const float* __restrict__ We,
    const float* __restrict__ V, const float* __restrict__ pbuf,
    const float* __restrict__ ssum, float* __restrict__ agg) {
  int t = blockIdx.x * 256 + threadIdx.x;
  if (t >= kE * kH) return;
  const int e  = t >> 2;
  const int hh = t & 3;
  const int sv = src[e];
  const int dv = dst[e];
  const float a = pbuf[t] / (ssum[dv * kH + hh] + 1e-16f);

  float ef[16];
  #pragma unroll
  for (int d = 0; d < 16; ++d) ef[d] = 0.0f;
  const float* ea = EA + (size_t)e * kED;
  #pragma unroll
  for (int j = 0; j < kED; ++j) {
    const float aj = ea[j];
    const float* wr = We + j * kHD + hh * 16;
    #pragma unroll
    for (int d = 0; d < 16; ++d) ef[d] += aj * wr[d];
  }
  const float* vr = V + (size_t)sv * kHD + hh * 16;
  float* ar = agg + (size_t)dv * kHD + hh * 16;
  #pragma unroll
  for (int d = 0; d < 16; ++d) atomicAdd(&ar[d], a * (vr[d] + ef[d]));
}

// ---------------------------------------------------------------------------
// beta gate + relu:  h = relu(beta*skip + (1-beta)*agg)
// ---------------------------------------------------------------------------
__global__ __launch_bounds__(256) void tc_beta_relu(
    const float* __restrict__ agg, const float* __restrict__ skip,
    const float* __restrict__ Wb, float* __restrict__ hout) {
  int n = blockIdx.x * 256 + threadIdx.x;
  if (n >= kN) return;
  const float* ar = agg + (size_t)n * kHD;
  const float* sr = skip + (size_t)n * kHD;
  float acc = 0.0f;
  #pragma unroll 8
  for (int j = 0; j < kHD; ++j) {
    const float a = ar[j], s = sr[j];
    acc += Wb[j] * a + Wb[kHD + j] * s + Wb[2 * kHD + j] * (a - s);
  }
  const float beta = 1.0f / (1.0f + expf(-acc));
  float* hr = hout + (size_t)n * kHD;
  #pragma unroll 8
  for (int j = 0; j < kHD; ++j) {
    const float a = ar[j], s = sr[j];
    hr[j] = fmaxf(0.0f, beta * s + (1.0f - beta) * a);
  }
}

// ---------------------------------------------------------------------------
// Pooling
// ---------------------------------------------------------------------------
__global__ __launch_bounds__(256) void tc_pool_init(unsigned* __restrict__ gmax_u,
                                                    float* __restrict__ gsum,
                                                    float* __restrict__ cnt) {
  int t = blockIdx.x * 256 + threadIdx.x;
  if (t < kG * kHD) { gmax_u[t] = kEncNegInf; gsum[t] = 0.0f; }
  if (t < kG) cnt[t] = 0.0f;
}

__global__ __launch_bounds__(256) void tc_pool(
    const float* __restrict__ h, const int* __restrict__ batch,
    unsigned* __restrict__ gmax_u, float* __restrict__ gsum,
    float* __restrict__ cnt) {
  int t = blockIdx.x * 256 + threadIdx.x;
  if (t >= kN * kHD) return;
  const int n = t / kHD;
  const int f = t - n * kHD;
  const int g = batch[n];
  const float v = h[t];
  atomicMax(&gmax_u[g * kHD + f], fenc(v));
  atomicAdd(&gsum[g * kHD + f], v);
  if (f == 0) atomicAdd(&cnt[g], 1.0f);
}

// ---------------------------------------------------------------------------
// MLP head: sigmoid( (concat[gmax, gmean] @ Wlin + blin) @ Wout + bout )
// one block per graph
// ---------------------------------------------------------------------------
__global__ __launch_bounds__(256) void tc_mlp(
    const unsigned* __restrict__ gmax_u, const float* __restrict__ gsum,
    const float* __restrict__ cnt, const float* __restrict__ Wlin,
    const float* __restrict__ blin, const float* __restrict__ Wout,
    const float* __restrict__ bout, float* __restrict__ out) {
  __shared__ float sfeat[2 * kHD];
  __shared__ float sred[kDN];
  const int g = blockIdx.x;
  const int t = threadIdx.x;
  const float c = cnt[g];
  if (t < kHD) {
    sfeat[t] = (c > 0.0f) ? fdec(gmax_u[g * kHD + t]) : 0.0f;
  } else if (t < 2 * kHD) {
    sfeat[t] = gsum[g * kHD + (t - kHD)] / fmaxf(c, 1.0f);
  }
  __syncthreads();
  float z = blin[t];
  #pragma unroll 8
  for (int j = 0; j < 2 * kHD; ++j) z += sfeat[j] * Wlin[j * kDN + t];
  sred[t] = z * Wout[t];
  __syncthreads();
  for (int s = kDN / 2; s > 0; s >>= 1) {
    if (t < s) sred[t] += sred[t + s];
    __syncthreads();
  }
  if (t == 0) out[g] = 1.0f / (1.0f + expf(-(sred[0] + bout[0])));
}

// ---------------------------------------------------------------------------
extern "C" void kernel_launch(void* const* d_in, const int* in_sizes, int n_in,
                              void* d_out, int out_size, void* d_ws, size_t ws_size,
                              hipStream_t stream) {
  (void)in_sizes; (void)n_in; (void)out_size; (void)ws_size;
  const float* x         = (const float*)d_in[0];
  const float* edge_attr = (const float*)d_in[1];
  const int*   eidx      = (const int*)d_in[2];
  const int*   batch     = (const int*)d_in[3];
  const int*   src = eidx;
  const int*   dst = eidx + kE;

  // workspace layout (floats)
  float* fws = (float*)d_ws;
  size_t o = 0;
  const size_t N64 = (size_t)kN * kHD;
  float* Q    = fws + o; o += N64;
  float* K    = fws + o; o += N64;
  float* V    = fws + o; o += N64;
  float* SK   = fws + o; o += N64;
  float* AGG  = fws + o; o += N64;
  float* Hb   = fws + o; o += N64;
  unsigned* M_U = (unsigned*)(fws + o); o += (size_t)kN * kH;
  float* SSUM = fws + o; o += (size_t)kN * kH;
  float* PBUF = fws + o; o += (size_t)kE * kH;
  unsigned* GMAX_U = (unsigned*)(fws + o); o += (size_t)kG * kHD;
  float* GSUM = fws + o; o += (size_t)kG * kHD;
  float* CNT  = fws + o; o += kG;

  const int gridInit  = (kN * kHD + 255) / 256;       // 12500
  const int gridGemm  = ((kN / 16) + 7) / 8;          // 391
  const int gridEdge  = (kE * kH + 255) / 256;        // 6250
  const int gridBeta  = (kN + 255) / 256;             // 196
  const int gridPoolI = (kG * kHD + 255) / 256;       // 32

  for (int layer = 0; layer < 4; ++layer) {
    const float *Wq_, *bq_, *Wk_, *bk_, *Wv_, *bv_, *We_, *Ws_, *bs_, *Wb_;
    const float* feat;
    if (layer == 0) {
      Wq_ = (const float*)d_in[4];  bq_ = (const float*)d_in[5];
      Wk_ = (const float*)d_in[6];  bk_ = (const float*)d_in[7];
      Wv_ = (const float*)d_in[8];  bv_ = (const float*)d_in[9];
      We_ = (const float*)d_in[10];
      Ws_ = (const float*)d_in[11]; bs_ = (const float*)d_in[12];
      Wb_ = (const float*)d_in[13];
      feat = x;
    } else {
      const size_t li = (size_t)(layer - 1);
      Wq_ = (const float*)d_in[14] + li * kHD * kHD;
      bq_ = (const float*)d_in[15] + li * kHD;
      Wk_ = (const float*)d_in[16] + li * kHD * kHD;
      bk_ = (const float*)d_in[17] + li * kHD;
      Wv_ = (const float*)d_in[18] + li * kHD * kHD;
      bv_ = (const float*)d_in[19] + li * kHD;
      We_ = (const float*)d_in[20] + li * kED * kHD;
      Ws_ = (const float*)d_in[21] + li * kHD * kHD;
      bs_ = (const float*)d_in[22] + li * kHD;
      Wb_ = (const float*)d_in[23] + li * 3 * kHD;
      feat = Hb;
    }

    tc_init_layer<<<gridInit, 256, 0, stream>>>(AGG, M_U, SSUM);
    tc_gemm_qkvs<<<gridGemm, 256, 0, stream>>>(feat, Wq_, bq_, Wk_, bk_,
                                               Wv_, bv_, Ws_, bs_, Q, K, V, SK);
    tc_edge_logits<<<gridEdge, 256, 0, stream>>>(edge_attr, src, dst, We_,
                                                 Q, K, PBUF, M_U);
    tc_edge_softmax<<<gridEdge, 256, 0, stream>>>(dst, M_U, PBUF, SSUM);
    tc_edge_agg<<<gridEdge, 256, 0, stream>>>(edge_attr, src, dst, We_,
                                              V, PBUF, SSUM, AGG);
    tc_beta_relu<<<gridBeta, 256, 0, stream>>>(AGG, SK, Wb_, Hb);
  }

  tc_pool_init<<<gridPoolI, 256, 0, stream>>>(GMAX_U, GSUM, CNT);
  tc_pool<<<gridInit, 256, 0, stream>>>(Hb, batch, GMAX_U, GSUM, CNT);
  tc_mlp<<<kG, 256, 0, stream>>>(GMAX_U, GSUM, CNT,
                                 (const float*)d_in[24], (const float*)d_in[25],
                                 (const float*)d_in[26], (const float*)d_in[27],
                                 (float*)d_out);
}
